// TransformerDecoderLayer_12962211299421
// MI455X (gfx1250) — matrix operand order repaired and checked by reading, not compiled, not run on hardware
//
#include <hip/hip_runtime.h>
#include <hip/hip_bf16.h>
#include <math.h>

typedef __bf16 bf16;
typedef __attribute__((ext_vector_type(16))) __bf16 v16bf;
typedef __attribute__((ext_vector_type(8)))  __bf16 v8bf;
typedef __attribute__((ext_vector_type(8)))  float  v8f;

// Problem constants (from reference)
constexpr int cB  = 2;
constexpr int cN  = 1024;
constexpr int cD  = 512;
constexpr int cNH = 8;
constexpr int cDH = 64;
constexpr int cC  = 256;
constexpr int cHW = 1024;
constexpr int cT  = 2048;   // B*N tokens
constexpr int cE  = 32;
constexpr int cK  = 4;
constexpr int cS  = 8;
constexpr int cR  = 64;
constexpr int cGD = 64;
constexpr int cDFF= 512;

__device__ inline float gelu_tanh(float x) {
  float x3 = x * x * x;
  return 0.5f * x * (1.0f + tanhf(0.7978845608028654f * (x + 0.044715f * x3)));
}

// ---------------------------------------------------------------------------
// Generic batched WMMA GEMM.  C[z] = alpha * A[z] @ B[z] (+bias)(+add1)(+add2)
// Tile: 128x128x32, 256 threads = 8 waves (2x4), each wave 64x32 = 4x2 frags.
// OMODE: 0 = f32 plain (+bias,+add1,+add2)
//        1 = bf16 plain (+bias)
//        2 = bf16 scatter to [B,NH,N,dh]   (q / v head layout)
//        3 = bf16 scatter to [B,NH,dh,N]   (k transposed head layout)
//        4 = f32 atomicAdd into C[r*ldc+c] (+bias)   (shared-expert accum)
// ---------------------------------------------------------------------------
struct GArg {
  const bf16* A; const bf16* B; const float* bias;
  const float* add1; const float* add2;
  void* C;
  int M, N, K;
  int lda, ldb, ldc;
  int zdiv;
  long sAo, sAi, sBo, sBi, sCo, sCi, sBias;
  float alpha;
};

template<int OMODE, int GELU>
__global__ __launch_bounds__(256) void gemm_k(GArg g) {
  __shared__ bf16 As[128][40];
  __shared__ bf16 Bt[128][40];

  const int tid  = threadIdx.x;
  const int lane = tid & 31;
  const int wv   = tid >> 5;
  const int wm   = wv >> 2;      // 0..1
  const int wn   = wv & 3;       // 0..3
  const int hi   = lane >> 4;    // 0..1
  const int lo16 = lane & 15;

  const int z    = blockIdx.z;
  const int row0 = blockIdx.y * 128;
  const int col0 = blockIdx.x * 128;
  const long zo = z / g.zdiv, zi = z % g.zdiv;
  const bf16* Ag = g.A + zo * g.sAo + zi * g.sAi;
  const bf16* Bg = g.B + zo * g.sBo + zi * g.sBi;

  v8f acc[4][2];
  #pragma unroll
  for (int a = 0; a < 4; ++a)
    #pragma unroll
    for (int b = 0; b < 2; ++b)
      #pragma unroll
      for (int v = 0; v < 8; ++v) acc[a][b][v] = 0.0f;

  const int ar = tid >> 1;            // A fill: row 0..127
  const int ak = (tid & 1) * 16;      // A fill: k 0 or 16
  const int bk = tid >> 3;            // B fill: k 0..31
  const int bn = (tid & 7) * 16;      // B fill: n base 0..112

  for (int kt = 0; kt < g.K; kt += 32) {
    // ---- fill A tile [128 x 32] ----
    {
      const int gr = row0 + ar;
      v8bf a0 = {}, a1 = {};
      if (gr < g.M) {
        const bf16* src = Ag + (long)gr * g.lda + kt + ak;
        a0 = *(const v8bf*)(src);
        a1 = *(const v8bf*)(src + 8);
        if (kt + 32 < g.K)
          __builtin_prefetch(src + 32, 0, 1);   // global_prefetch_b8 next K-tile
      }
      *(v8bf*)&As[ar][ak]     = a0;
      *(v8bf*)&As[ar][ak + 8] = a1;
    }
    // ---- fill B tile transposed: Bt[n][k] = B[kt+k][col0+n] ----
    {
      const bf16* src = Bg + (long)(kt + bk) * g.ldb + col0 + bn;
      if (col0 + bn + 15 < g.N) {
        if (kt + 32 < g.K) __builtin_prefetch(src + 32L * g.ldb, 0, 1);
        #pragma unroll
        for (int i = 0; i < 16; ++i) Bt[bn + i][bk] = src[i];
      } else {
        #pragma unroll
        for (int i = 0; i < 16; ++i) {
          bf16 v = (col0 + bn + i < g.N) ? src[i] : (bf16)0.0f;
          Bt[bn + i][bk] = v;
        }
      }
    }
    __syncthreads();

    // ---- build fragments & WMMA ----
    v16bf bfr[2];
    #pragma unroll
    for (int fn = 0; fn < 2; ++fn) {
      const bf16* p = &Bt[wn * 32 + fn * 16 + lo16][0];
      v8bf b0 = *(const v8bf*)(p + hi * 8);
      v8bf b1 = *(const v8bf*)(p + 16 + hi * 8);
      #pragma unroll
      for (int i = 0; i < 8; ++i) { bfr[fn][i] = b0[i]; bfr[fn][i + 8] = b1[i]; }
    }
    v16bf afr[4];
    #pragma unroll
    for (int fm = 0; fm < 4; ++fm) {
      const bf16* p = &As[wm * 64 + fm * 16 + lo16][0];
      v8bf a0 = *(const v8bf*)(p + hi * 8);
      v8bf a1 = *(const v8bf*)(p + 16 + hi * 8);
      #pragma unroll
      for (int i = 0; i < 8; ++i) { afr[fm][i] = a0[i]; afr[fm][i + 8] = a1[i]; }
    }
    #pragma unroll
    for (int fm = 0; fm < 4; ++fm)
      #pragma unroll
      for (int fn = 0; fn < 2; ++fn)
        acc[fm][fn] = __builtin_amdgcn_wmma_f32_16x16x32_bf16(
            false, afr[fm], false, bfr[fn], (short)0, acc[fm][fn], false, false);
    __syncthreads();
  }

  // ---- epilogue ----
  const long co = zo * g.sCo + zi * g.sCi;
  #pragma unroll
  for (int fm = 0; fm < 4; ++fm) {
    #pragma unroll
    for (int fn = 0; fn < 2; ++fn) {
      #pragma unroll
      for (int v = 0; v < 8; ++v) {
        const int r = row0 + wm * 64 + fm * 16 + hi * 8 + v;
        const int c = col0 + wn * 32 + fn * 16 + lo16;
        if (r < g.M && c < g.N) {
          float val = g.alpha * acc[fm][fn][v];
          if (g.bias) val += g.bias[(long)z * g.sBias + c];
          if (GELU) val = gelu_tanh(val);
          if (OMODE == 0) {
            if (g.add1) val += g.add1[(long)r * g.ldc + c];
            if (g.add2) val += g.add2[(long)r * g.ldc + c];
            ((float*)g.C)[co + (long)r * g.ldc + c] = val;
          } else if (OMODE == 1) {
            ((bf16*)g.C)[co + (long)r * g.ldc + c] = (bf16)val;
          } else if (OMODE == 2) {
            // [T, D] -> [B, NH, N, dh]
            long d = ((((long)(r >> 10) * cNH + (c >> 6)) * cN) + (r & 1023)) * cDH + (c & 63);
            ((bf16*)g.C)[d] = (bf16)val;
          } else if (OMODE == 3) {
            // [T, D] -> [B, NH, dh, N]
            long d = ((((long)(r >> 10) * cNH + (c >> 6)) * cDH) + (c & 63)) * cN + (r & 1023);
            ((bf16*)g.C)[d] = (bf16)val;
          } else { // 4
            atomicAdd(((float*)g.C) + (long)r * g.ldc + c, val);
          }
        }
      }
    }
  }
}

// ---------------------------------------------------------------------------
// Grouped MoE GEMM (per-expert token groups, 512x512 weights, K=N=512).
// PHASE 1: H[g] = gelu(X[gidx[g]] @ W[e] + b1[e])        (bf16 out)
// PHASE 2: atomicAdd(Y[gidx[g]] , gwt[g]*(H[g]@W2[e]+b2[e]))
// ---------------------------------------------------------------------------
struct GrpArg {
  const bf16* X; const bf16* W; const float* bias;
  const int* offs; const int* gidx; const float* gwt;
  bf16* H; float* Y;
};

template<int PHASE>
__global__ __launch_bounds__(256) void moe_gemm_k(GrpArg p) {
  const int z  = blockIdx.z;
  const int g0 = p.offs[z];
  const int nrows = p.offs[z + 1] - g0;
  const int row0 = blockIdx.y * 128;
  if (row0 >= nrows) return;

  __shared__ bf16 As[128][40];
  __shared__ bf16 Bt[128][40];

  const int tid  = threadIdx.x;
  const int lane = tid & 31;
  const int wv   = tid >> 5;
  const int wm   = wv >> 2, wn = wv & 3;
  const int hi   = lane >> 4, lo16 = lane & 15;
  const int col0 = blockIdx.x * 128;
  const bf16* Wg = p.W + (long)z * cD * cDFF;

  v8f acc[4][2];
  #pragma unroll
  for (int a = 0; a < 4; ++a)
    #pragma unroll
    for (int b = 0; b < 2; ++b)
      #pragma unroll
      for (int v = 0; v < 8; ++v) acc[a][b][v] = 0.0f;

  const int ar = tid >> 1;
  const int ak = (tid & 1) * 16;
  const int bk = tid >> 3;
  const int bn = (tid & 7) * 16;

  long srow = -1;
  if (row0 + ar < nrows)
    srow = (PHASE == 1) ? (long)p.gidx[g0 + row0 + ar] : (long)(g0 + row0 + ar);

  for (int kt = 0; kt < cD; kt += 32) {
    {
      v8bf a0 = {}, a1 = {};
      if (srow >= 0) {
        const bf16* src = p.X + srow * cD + kt + ak;
        a0 = *(const v8bf*)(src);
        a1 = *(const v8bf*)(src + 8);
      }
      *(v8bf*)&As[ar][ak]     = a0;
      *(v8bf*)&As[ar][ak + 8] = a1;
    }
    {
      const bf16* src = Wg + (long)(kt + bk) * cDFF + col0 + bn;
      #pragma unroll
      for (int i = 0; i < 16; ++i) Bt[bn + i][bk] = src[i];
    }
    __syncthreads();

    v16bf bfr[2];
    #pragma unroll
    for (int fn = 0; fn < 2; ++fn) {
      const bf16* q = &Bt[wn * 32 + fn * 16 + lo16][0];
      v8bf b0 = *(const v8bf*)(q + hi * 8);
      v8bf b1 = *(const v8bf*)(q + 16 + hi * 8);
      #pragma unroll
      for (int i = 0; i < 8; ++i) { bfr[fn][i] = b0[i]; bfr[fn][i + 8] = b1[i]; }
    }
    v16bf afr[4];
    #pragma unroll
    for (int fm = 0; fm < 4; ++fm) {
      const bf16* q = &As[wm * 64 + fm * 16 + lo16][0];
      v8bf a0 = *(const v8bf*)(q + hi * 8);
      v8bf a1 = *(const v8bf*)(q + 16 + hi * 8);
      #pragma unroll
      for (int i = 0; i < 8; ++i) { afr[fm][i] = a0[i]; afr[fm][i + 8] = a1[i]; }
    }
    #pragma unroll
    for (int fm = 0; fm < 4; ++fm)
      #pragma unroll
      for (int fn = 0; fn < 2; ++fn)
        acc[fm][fn] = __builtin_amdgcn_wmma_f32_16x16x32_bf16(
            false, afr[fm], false, bfr[fn], (short)0, acc[fm][fn], false, false);
    __syncthreads();
  }

  #pragma unroll
  for (int fm = 0; fm < 4; ++fm) {
    #pragma unroll
    for (int fn = 0; fn < 2; ++fn) {
      #pragma unroll
      for (int v = 0; v < 8; ++v) {
        const int r = row0 + wm * 64 + fm * 16 + hi * 8 + v;
        const int c = col0 + wn * 32 + fn * 16 + lo16;
        if (r < nrows && c < cDFF) {
          float val = acc[fm][fn][v] + p.bias[(long)z * cDFF + c];
          if (PHASE == 1) {
            p.H[(long)(g0 + r) * cDFF + c] = (bf16)gelu_tanh(val);
          } else {
            long tok = (long)p.gidx[g0 + r];
            atomicAdd(p.Y + tok * cD + c, p.gwt[g0 + r] * val);
          }
        }
      }
    }
  }
}

// ---------------------------------------------------------------------------
// Small helper kernels
// ---------------------------------------------------------------------------
__global__ void cvt_k(const float* in, bf16* out, long n) {
  long i = (long)blockIdx.x * 256 + threadIdx.x;
  if (i < n) out[i] = (bf16)in[i];
}
__global__ void zerof_k(float* p, long n) {
  long i = (long)blockIdx.x * 256 + threadIdx.x;
  if (i < n) p[i] = 0.0f;
}
__global__ void zeroi_k(int* p, long n) {
  long i = (long)blockIdx.x * 256 + threadIdx.x;
  if (i < n) p[i] = 0;
}
__global__ void cnnt_k(const float* cnn, bf16* out) {
  long i = (long)blockIdx.x * 256 + threadIdx.x;   // over B*HW*C
  if (i >= (long)cB * cHW * cC) return;
  int c = (int)(i % cC);
  long rest = i / cC;
  int hw = (int)(rest % cHW);
  int b  = (int)(rest / cHW);
  out[i] = (bf16)cnn[((long)b * cC + c) * cHW + hw];
}
// LayerNorm over D=512, one block (128 threads) per token; bf16 (+optional f32) out
__global__ void ln_k(const float* x, const float* gw, const float* bw,
                     bf16* ybf, float* yf) {
  const int t = blockIdx.x;
  const int tid = threadIdx.x;
  const float* xr = x + (long)t * cD;
  __shared__ float red[4];
  float s = 0.0f;
  for (int i = tid; i < cD; i += 128) s += xr[i];
  #pragma unroll
  for (int o = 16; o > 0; o >>= 1) s += __shfl_xor(s, o, 32);
  if ((tid & 31) == 0) red[tid >> 5] = s;
  __syncthreads();
  const float mean = (red[0] + red[1] + red[2] + red[3]) * (1.0f / cD);
  __syncthreads();
  float v = 0.0f;
  for (int i = tid; i < cD; i += 128) { float d = xr[i] - mean; v += d * d; }
  #pragma unroll
  for (int o = 16; o > 0; o >>= 1) v += __shfl_xor(v, o, 32);
  if ((tid & 31) == 0) red[tid >> 5] = v;
  __syncthreads();
  const float var = (red[0] + red[1] + red[2] + red[3]) * (1.0f / cD);
  const float rs = rsqrtf(var + 1e-5f);
  for (int i = tid; i < cD; i += 128) {
    float o = (xr[i] - mean) * rs * gw[i] + bw[i];
    ybf[(long)t * cD + i] = (bf16)o;
    if (yf) yf[(long)t * cD + i] = o;
  }
}
// Row softmax over nc columns, bf16 out; one block (256 threads) per row
__global__ void softmax_k(const float* S, bf16* P, int nc) {
  const long row = blockIdx.x;
  const float* sr = S + row * (long)nc;
  const int tid = threadIdx.x;
  __shared__ float red[8];
  float mx = -1e30f;
  for (int i = tid; i < nc; i += 256) mx = fmaxf(mx, sr[i]);
  #pragma unroll
  for (int o = 16; o > 0; o >>= 1) mx = fmaxf(mx, __shfl_xor(mx, o, 32));
  if ((tid & 31) == 0) red[tid >> 5] = mx;
  __syncthreads();
  mx = red[0];
  #pragma unroll
  for (int i = 1; i < 8; ++i) mx = fmaxf(mx, red[i]);
  __syncthreads();
  float sum = 0.0f;
  for (int i = tid; i < nc; i += 256) sum += __expf(sr[i] - mx);
  #pragma unroll
  for (int o = 16; o > 0; o >>= 1) sum += __shfl_xor(sum, o, 32);
  if ((tid & 31) == 0) red[tid >> 5] = sum;
  __syncthreads();
  sum = red[0] + red[1] + red[2] + red[3] + red[4] + red[5] + red[6] + red[7];
  const float inv = 1.0f / sum;
  for (int i = tid; i < nc; i += 256)
    P[row * (long)nc + i] = (bf16)(__expf(sr[i] - mx) * inv);
}
// Router softmax + top-k(4) + stats for aux loss
__global__ void topk_k(const float* logits, int* idx, float* wt,
                       int* cnt, float* sumP) {
  const int t = blockIdx.x * blockDim.x + threadIdx.x;
  if (t >= cT) return;
  float p[cE];
  const float* lr = logits + (long)t * cE;
  float mx = -1e30f;
  #pragma unroll
  for (int e = 0; e < cE; ++e) { p[e] = lr[e]; mx = fmaxf(mx, p[e]); }
  float s = 0.0f;
  #pragma unroll
  for (int e = 0; e < cE; ++e) { p[e] = __expf(p[e] - mx); s += p[e]; }
  const float inv = 1.0f / s;
  #pragma unroll
  for (int e = 0; e < cE; ++e) { p[e] *= inv; atomicAdd(&sumP[e], p[e]); }
  bool used[cE] = {};
  int id[cK]; float vv[cK]; float vsum = 0.0f;
  for (int k = 0; k < cK; ++k) {
    int bi = 0; float bv = -1.0f;
    for (int e = 0; e < cE; ++e)
      if (!used[e] && p[e] > bv) { bv = p[e]; bi = e; }
    used[bi] = true; id[k] = bi; vv[k] = bv; vsum += bv;
  }
  const float wn = 1.0f / vsum;
  for (int k = 0; k < cK; ++k) {
    idx[t * cK + k] = id[k];
    wt[t * cK + k]  = vv[k] * wn;
    atomicAdd(&cnt[id[k]], 1);
  }
}
__global__ void scan_k(const int* cnt, int* offs, int* cursor) {
  if (threadIdx.x == 0) {
    int acc = 0;
    for (int e = 0; e < cE; ++e) { offs[e] = acc; cursor[e] = acc; acc += cnt[e]; }
    offs[cE] = acc;
  }
}
__global__ void scatter_k(const int* idx, const float* wt, int* cursor,
                          int* gidx, float* gwt) {
  const int t = blockIdx.x * blockDim.x + threadIdx.x;
  if (t >= cT) return;
  for (int k = 0; k < cK; ++k) {
    int e = idx[t * cK + k];
    int pos = atomicAdd(&cursor[e], 1);
    gidx[pos] = t;
    gwt[pos]  = wt[t * cK + k];
  }
}
__global__ void aux_k(const int* cnt, const float* sumP, float* out) {
  const int e = threadIdx.x;   // 32 threads
  float f  = (float)cnt[e] / (float)(cT * cK);
  float mp = sumP[e] / (float)cT;
  float v = f * mp;
  #pragma unroll
  for (int o = 16; o > 0; o >>= 1) v += __shfl_xor(v, o, 32);
  if (e == 0) out[0] = (float)cE * v;
}
__global__ void addcvt_k(const float* a, const float* b, float* o, bf16* obf, long n) {
  long i = (long)blockIdx.x * 256 + threadIdx.x;
  if (i < n) { float v = a[i] + b[i]; o[i] = v; obf[i] = (bf16)v; }
}
__global__ void meanN_k(const float* x, float* out) {
  const int i = blockIdx.x * blockDim.x + threadIdx.x;   // over B*D
  if (i >= cB * cD) return;
  const int b = i / cD, d = i % cD;
  float s = 0.0f;
  for (int n = 0; n < cN; ++n) s += x[((long)b * cN + n) * cD + d];
  out[i] = s * (1.0f / cN);
}
__global__ void gate_k(const float* xmean, const float* gWg, float* out) {
  const int i = threadIdx.x;   // 128 threads, B*GD outputs
  if (i >= cB * cGD) return;
  const int b = i / cGD, g = i % cGD;
  float s = 0.0f;
  for (int d = 0; d < cD; ++d) s += xmean[b * cD + d] * gWg[(long)d * cGD + g];
  out[i] = 1.0f / (1.0f + __expf(-s));
}

// ---------------------------------------------------------------------------
// Host orchestration
// ---------------------------------------------------------------------------
template<int OM, int GELU>
static void LG(hipStream_t s, const GArg& g, int zb) {
  dim3 grid((g.N + 127) / 128, (g.M + 127) / 128, zb);
  gemm_k<OM, GELU><<<grid, dim3(256), 0, s>>>(g);
}

static GArg baseG() {
  GArg g{};
  g.zdiv = 1; g.alpha = 1.0f;
  g.sAo = g.sAi = g.sBo = g.sBi = g.sCo = g.sCi = g.sBias = 0;
  return g;
}

extern "C" void kernel_launch(void* const* d_in, const int* in_sizes, int n_in,
                              void* d_out, int out_size, void* d_ws, size_t ws_size,
                              hipStream_t stream) {
  const float* x_in  = (const float*)d_in[0];
  const float* cnn   = (const float*)d_in[1];
  const float* Wc    = (const float*)d_in[2];
  const float* bc    = (const float*)d_in[3];
  const float* lnq_g = (const float*)d_in[4];
  const float* lnq_b = (const float*)d_in[5];
  const float* lnk_g = (const float*)d_in[6];
  const float* lnk_b = (const float*)d_in[7];
  const float* lnv_g = (const float*)d_in[8];
  const float* lnv_b = (const float*)d_in[9];
  const float* ln1_g = (const float*)d_in[10];
  const float* ln1_b = (const float*)d_in[11];
  const float* ln2_g = (const float*)d_in[12];
  const float* ln2_b = (const float*)d_in[13];
  const float* saW[5] = {(const float*)d_in[14], (const float*)d_in[15],
                         (const float*)d_in[16], (const float*)d_in[17],
                         (const float*)d_in[18]};
  const float* sab[5] = {(const float*)d_in[19], (const float*)d_in[20],
                         (const float*)d_in[21], (const float*)d_in[22],
                         (const float*)d_in[23]};
  const float* caW[5] = {(const float*)d_in[24], (const float*)d_in[25],
                         (const float*)d_in[26], (const float*)d_in[27],
                         (const float*)d_in[28]};
  const float* cab[5] = {(const float*)d_in[29], (const float*)d_in[30],
                         (const float*)d_in[31], (const float*)d_in[32],
                         (const float*)d_in[33]};
  const float* rW1 = (const float*)d_in[34];
  const float* rW2 = (const float*)d_in[35];
  const float* eW1 = (const float*)d_in[36];
  const float* eb1 = (const float*)d_in[37];
  const float* eW2 = (const float*)d_in[38];
  const float* eb2 = (const float*)d_in[39];
  const float* sW1 = (const float*)d_in[40];
  const float* sb1 = (const float*)d_in[41];
  const float* sW2 = (const float*)d_in[42];
  const float* sb2 = (const float*)d_in[43];
  const float* gWg = (const float*)d_in[44];
  const float* gWf = (const float*)d_in[45];

  char* wp = (char*)d_ws;
  auto alloc = [&](size_t bytes) -> void* {
    void* r = wp; wp += (bytes + 255) & ~(size_t)255; return r;
  };
  auto cvt = [&](const float* src, size_t n) -> bf16* {
    bf16* dst = (bf16*)alloc(n * sizeof(bf16));
    cvt_k<<<dim3((unsigned)((n + 255) / 256)), dim3(256), 0, stream>>>(src, dst, (long)n);
    return dst;
  };

  // ---- weights -> bf16 ----
  bf16* Wcb  = cvt(Wc, (size_t)cC * cD);
  bf16* saWb[5]; for (int i = 0; i < 5; ++i) saWb[i] = cvt(saW[i], (size_t)cD * cD);
  bf16* caWb[5]; for (int i = 0; i < 5; ++i) caWb[i] = cvt(caW[i], (size_t)cD * cD);
  bf16* rW1b = cvt(rW1, (size_t)cD * cR);
  bf16* rW2b = cvt(rW2, (size_t)cR * cE);
  bf16* eW1b = cvt(eW1, (size_t)cE * cD * cDFF);
  bf16* eW2b = cvt(eW2, (size_t)cE * cDFF * cD);
  bf16* sW1b = cvt(sW1, (size_t)cS * cD * cDFF);
  bf16* sW2b = cvt(sW2, (size_t)cS * cDFF * cD);
  bf16* gWfb = cvt(gWf, (size_t)cD * cD);

  // ---- activations / scratch ----
  bf16*  cnnTb  = (bf16*)alloc((size_t)cT * cC * 2);
  float* cnnTok = (float*)alloc((size_t)cT * cD * 4);
  bf16*  kkb    = (bf16*)alloc((size_t)cT * cD * 2);
  bf16*  vvb    = (bf16*)alloc((size_t)cT * cD * 2);
  bf16*  xlnq   = (bf16*)alloc((size_t)cT * cD * 2);
  float* sbr    = (float*)alloc((size_t)cT * cD * 4);
  bf16*  qh     = (bf16*)alloc((size_t)cT * cD * 2);
  bf16*  khT    = (bf16*)alloc((size_t)cT * cD * 2);
  bf16*  vh     = (bf16*)alloc((size_t)cT * cD * 2);
  float* scores = (float*)alloc((size_t)cB * cNH * cN * cN * 4);
  bf16*  att    = (bf16*)alloc((size_t)cB * cNH * cN * cN * 2);
  bf16*  oc     = (bf16*)alloc((size_t)cT * cD * 2);
  float* x1     = (float*)alloc((size_t)cT * cD * 4);
  float* res2   = (float*)alloc((size_t)cT * cD * 4);
  bf16*  res2b  = (bf16*)alloc((size_t)cT * cD * 2);
  float* x2     = (float*)alloc((size_t)cT * cD * 4);
  float* res3   = (float*)alloc((size_t)cT * cD * 4);
  bf16*  xm     = (bf16*)alloc((size_t)cT * cD * 2);
  bf16*  r1b    = (bf16*)alloc((size_t)cT * cR * 2);
  float* rlog   = (float*)alloc((size_t)cT * cE * 4);
  int*   tidx   = (int*)alloc((size_t)cT * cK * 4);
  float* twt    = (float*)alloc((size_t)cT * cK * 4);
  int*   cnt    = (int*)alloc(cE * 4);
  float* sumP   = (float*)alloc(cE * 4);
  int*   offs   = (int*)alloc((cE + 1) * 4);
  int*   cursor = (int*)alloc(cE * 4);
  int*   gidx   = (int*)alloc((size_t)cT * cK * 4);
  float* gwt    = (float*)alloc((size_t)cT * cK * 4);
  bf16*  hbuf   = (bf16*)alloc((size_t)cT * cK * cDFF * 2);
  bf16*  hsb    = (bf16*)alloc((size_t)cS * cT * cDFF * 2);
  float* moeout = (float*)alloc((size_t)cT * cD * 4);
  float* x3     = (float*)alloc((size_t)cT * cD * 4);
  bf16*  x3b    = (bf16*)alloc((size_t)cT * cD * 2);
  float* xmean  = (float*)alloc((size_t)cB * cD * 4);

  float* out_query = (float*)d_out;                      // [2,1024,512]
  float* out_aux   = (float*)d_out + (size_t)cT * cD;    // [1]
  float* out_gate  = out_aux + 1;                        // [2,64]

  const int nheads = cB * cNH;

  // =================== Self-attention block ===================
  ln_k<<<dim3(cT), dim3(128), 0, stream>>>(x_in, lnq_g, lnq_b, xlnq, nullptr);
  {
    GArg g = baseG();
    g.A = xlnq; g.M = cT; g.K = cD; g.N = cD; g.lda = cD; g.ldb = cD; g.ldc = cD;
    g.B = saWb[0]; g.bias = sab[0]; g.C = qh;  LG<2,0>(stream, g, 1);
    g.B = saWb[1]; g.bias = sab[1]; g.C = khT; LG<3,0>(stream, g, 1);
    g.B = saWb[2]; g.bias = sab[2]; g.C = vh;  LG<2,0>(stream, g, 1);
    g.B = saWb[4]; g.bias = sab[4]; g.C = sbr; LG<0,0>(stream, g, 1);
  }
  { // scores = (qh @ khT) / 8  per (b,h)
    GArg g = baseG();
    g.A = qh; g.B = khT; g.C = scores;
    g.M = cN; g.N = cN; g.K = cDH; g.lda = cDH; g.ldb = cN; g.ldc = cN;
    g.sAo = (long)cN * cDH; g.sBo = (long)cDH * cN; g.sCo = (long)cN * cN;
    g.alpha = 0.125f;
    LG<0,0>(stream, g, nheads);
  }
  softmax_k<<<dim3(nheads * cN), dim3(256), 0, stream>>>(scores, att, cN);
  { // oc[b,n, h*64+dd] = att @ vh
    GArg g = baseG();
    g.A = att; g.B = vh; g.C = oc;
    g.M = cN; g.N = cDH; g.K = cN; g.lda = cN; g.ldb = cDH; g.ldc = cD;
    g.zdiv = cNH;
    g.sAo = (long)cNH * cN * cN; g.sAi = (long)cN * cN;
    g.sBo = (long)cNH * cN * cDH; g.sBi = (long)cN * cDH;
    g.sCo = (long)cN * cD;        g.sCi = cDH;
    LG<1,0>(stream, g, nheads);
  }
  { // x1 = x_in + oc@Wo + bo + sbr
    GArg g = baseG();
    g.A = oc; g.B = saWb[3]; g.bias = sab[3]; g.add1 = sbr; g.add2 = x_in;
    g.C = x1; g.M = cT; g.N = cD; g.K = cD; g.lda = cD; g.ldb = cD; g.ldc = cD;
    LG<0,0>(stream, g, 1);
  }

  // =================== Cross-attention block ===================
  ln_k<<<dim3(cT), dim3(128), 0, stream>>>(x1, ln1_g, ln1_b, res2b, res2);
  cnnt_k<<<dim3((cT * cC + 255) / 256), dim3(256), 0, stream>>>(cnn, cnnTb);
  { // cnn_tokens = cnnT @ Wc + bc
    GArg g = baseG();
    g.A = cnnTb; g.B = Wcb; g.bias = bc; g.C = cnnTok;
    g.M = cT; g.N = cD; g.K = cC; g.lda = cC; g.ldb = cD; g.ldc = cD;
    LG<0,0>(stream, g, 1);
  }
  ln_k<<<dim3(cT), dim3(128), 0, stream>>>(cnnTok, lnk_g, lnk_b, kkb, nullptr);
  ln_k<<<dim3(cT), dim3(128), 0, stream>>>(cnnTok, lnv_g, lnv_b, vvb, nullptr);
  {
    GArg g = baseG();
    g.M = cT; g.K = cD; g.N = cD; g.lda = cD; g.ldb = cD; g.ldc = cD;
    g.A = res2b; g.B = caWb[0]; g.bias = cab[0]; g.C = qh;  LG<2,0>(stream, g, 1);
    g.A = kkb;   g.B = caWb[1]; g.bias = cab[1]; g.C = khT; LG<3,0>(stream, g, 1);
    g.A = vvb;   g.B = caWb[2]; g.bias = cab[2]; g.C = vh;  LG<2,0>(stream, g, 1);
    g.A = res2b; g.B = caWb[4]; g.bias = cab[4]; g.C = sbr; LG<0,0>(stream, g, 1);
  }
  {
    GArg g = baseG();
    g.A = qh; g.B = khT; g.C = scores;
    g.M = cN; g.N = cN; g.K = cDH; g.lda = cDH; g.ldb = cN; g.ldc = cN;
    g.sAo = (long)cN * cDH; g.sBo = (long)cDH * cN; g.sCo = (long)cN * cN;
    g.alpha = 0.125f;
    LG<0,0>(stream, g, nheads);
  }
  softmax_k<<<dim3(nheads * cN), dim3(256), 0, stream>>>(scores, att, cN);
  {
    GArg g = baseG();
    g.A = att; g.B = vh; g.C = oc;
    g.M = cN; g.N = cDH; g.K = cN; g.lda = cN; g.ldb = cDH; g.ldc = cD;
    g.zdiv = cNH;
    g.sAo = (long)cNH * cN * cN; g.sAi = (long)cN * cN;
    g.sBo = (long)cNH * cN * cDH; g.sBi = (long)cN * cDH;
    g.sCo = (long)cN * cD;        g.sCi = cDH;
    LG<1,0>(stream, g, nheads);
  }
  {
    GArg g = baseG();
    g.A = oc; g.B = caWb[3]; g.bias = cab[3]; g.add1 = sbr; g.add2 = res2;
    g.C = x2; g.M = cT; g.N = cD; g.K = cD; g.lda = cD; g.ldb = cD; g.ldc = cD;
    LG<0,0>(stream, g, 1);
  }

  // =================== MoE block ===================
  ln_k<<<dim3(cT), dim3(128), 0, stream>>>(x2, ln2_g, ln2_b, xm, res3);
  { // router: r1 = xm @ rW1 (bf16), rlog = r1 @ rW2 (f32)
    GArg g = baseG();
    g.A = xm; g.B = rW1b; g.C = r1b;
    g.M = cT; g.N = cR; g.K = cD; g.lda = cD; g.ldb = cR; g.ldc = cR;
    LG<1,0>(stream, g, 1);
    GArg h = baseG();
    h.A = r1b; h.B = rW2b; h.C = rlog;
    h.M = cT; h.N = cE; h.K = cR; h.lda = cR; h.ldb = cE; h.ldc = cE;
    LG<0,0>(stream, h, 1);
  }
  zeroi_k<<<dim3(1), dim3(256), 0, stream>>>(cnt, cE);
  zerof_k<<<dim3(1), dim3(256), 0, stream>>>(sumP, cE);
  zerof_k<<<dim3(((long)cT * cD + 255) / 256), dim3(256), 0, stream>>>(moeout, (long)cT * cD);
  topk_k<<<dim3((cT + 255) / 256), dim3(256), 0, stream>>>(rlog, tidx, twt, cnt, sumP);
  scan_k<<<dim3(1), dim3(32), 0, stream>>>(cnt, offs, cursor);
  scatter_k<<<dim3((cT + 255) / 256), dim3(256), 0, stream>>>(tidx, twt, cursor, gidx, gwt);
  { // routed experts: grouped gather-GEMMs
    GrpArg p{};
    p.X = xm; p.W = eW1b; p.bias = eb1; p.offs = offs; p.gidx = gidx; p.gwt = gwt;
    p.H = hbuf; p.Y = moeout;
    dim3 grid(cDFF / 128, cT / 128, cE);
    moe_gemm_k<1><<<grid, dim3(256), 0, stream>>>(p);
    GrpArg q = p;
    q.X = hbuf; q.W = eW2b; q.bias = eb2;
    moe_gemm_k<2><<<grid, dim3(256), 0, stream>>>(q);
  }
  { // shared experts: dense batched over S=8, accumulate into moeout
    GArg g = baseG();
    g.A = xm; g.B = sW1b; g.bias = sb1; g.C = hsb;
    g.M = cT; g.N = cDFF; g.K = cD; g.lda = cD; g.ldb = cDFF; g.ldc = cDFF;
    g.sAo = 0; g.sBo = (long)cD * cDFF; g.sCo = (long)cT * cDFF; g.sBias = cDFF;
    LG<1,1>(stream, g, cS);                 // bf16 out + GELU
    GArg h = baseG();
    h.A = hsb; h.B = sW2b; h.bias = sb2; h.C = moeout;
    h.M = cT; h.N = cD; h.K = cDFF; h.lda = cDFF; h.ldb = cD; h.ldc = cD;
    h.sAo = (long)cT * cDFF; h.sBo = (long)cDFF * cD; h.sBias = cD;
    LG<4,0>(stream, h, cS);                 // atomic f32 accumulate
  }
  addcvt_k<<<dim3(((long)cT * cD + 255) / 256), dim3(256), 0, stream>>>(
      res3, moeout, x3, x3b, (long)cT * cD);
  aux_k<<<dim3(1), dim3(32), 0, stream>>>(cnt, sumP, out_aux);

  // =================== Outputs ===================
  { // query = x3 @ gWf
    GArg g = baseG();
    g.A = x3b; g.B = gWfb; g.C = out_query;
    g.M = cT; g.N = cD; g.K = cD; g.lda = cD; g.ldb = cD; g.ldc = cD;
    LG<0,0>(stream, g, 1);
  }
  meanN_k<<<dim3((cB * cD + 255) / 256), dim3(256), 0, stream>>>(x3, xmean);
  gate_k<<<dim3(1), dim3(128), 0, stream>>>(xmean, gWg, out_gate);
}
